// MultiHeadedSelfAttention_21328807592161
// MI455X (gfx1250) — matrix operand order repaired and checked by reading, not compiled, not run on hardware
//
#include <hip/hip_runtime.h>

// ---------------------------------------------------------------------------
// MHSA for MI455X (gfx1250, wave32, WMMA 16x16x32 bf16, async-to-LDS copies).
// Phase 0: convert q/k/v and Wq/Wk/Wv fp32 -> bf16 (one streaming pass).
// Phase 1: QKV projections (bf16 GEMM, x @ W^T + b) -> bf16 workspace.
//          Q,K stored [B,H,S,DH]; V stored transposed [B,H,DH,S].
// Phase 2: fused flash-attention, K/V blocks async-staged in LDS,
//          fp32 accumulate, fp32 output [B,S,D].
// ---------------------------------------------------------------------------

typedef __attribute__((ext_vector_type(16))) __bf16 bf16x16;
typedef __attribute__((ext_vector_type(8)))  __bf16 bf16x8;
typedef __attribute__((ext_vector_type(4)))  __bf16 bf16x4;
typedef __attribute__((ext_vector_type(8)))  float  f32x8;

union Frag {
    bf16x16 f;
    bf16x8  h[2];
};

__device__ __forceinline__ f32x8 wmma_bf16(bf16x16 a, bf16x16 b, f32x8 c) {
    return __builtin_amdgcn_wmma_f32_16x16x32_bf16(
        /*neg_a=*/false, a, /*neg_b=*/false, b,
        /*c_mod=*/(short)0, c, /*reuse_a=*/false, /*reuse_b=*/false);
}

// Async 16B global -> LDS copy (ASYNCcnt). VDST vgpr holds LDS byte offset,
// address pair holds the 64-bit global address.
__device__ __forceinline__ void async_copy16(const void* gptr, void* lptr) {
    unsigned lds_off = (unsigned)(unsigned long long)lptr;  // AS3 offset = low 32 bits
    unsigned long long ga = (unsigned long long)gptr;
    asm volatile("global_load_async_to_lds_b128 %0, %1, off"
                 :: "v"(lds_off), "v"(ga) : "memory");
}
#define WAIT_ASYNC(n) asm volatile("s_wait_asynccnt " #n ::: "memory")

static constexpr int S  = 1024;
static constexpr int D  = 1024;
static constexpr int H  = 16;
static constexpr int DH = 64;
static constexpr int LDT = 72;  // LDS tile row stride in bf16 (144B: 16B-aligned, no bank conflicts)

// ---------------------------------------------------------------------------
// Phase 0: fp32 -> bf16 streaming convert (vectorized 4-wide).
// ---------------------------------------------------------------------------
__global__ __launch_bounds__(256)
void cvt_f32_to_bf16_kernel(const float* __restrict__ in,
                            unsigned short* __restrict__ out, int n4) {
    int i = blockIdx.x * 256 + threadIdx.x;
    if (i < n4) {
        const float4 v = ((const float4*)in)[i];
        bf16x4 o;
        o[0] = (__bf16)v.x; o[1] = (__bf16)v.y;
        o[2] = (__bf16)v.z; o[3] = (__bf16)v.w;
        ((bf16x4*)out)[i] = o;
    }
}

// ---------------------------------------------------------------------------
// Phase 1: out[i,j] = sum_k xb[i,k]*Wb[j,k] + bias[j]; M=8192, N=1024, K=1024.
// Workgroup = 256 threads (8 waves), tile 128(M) x 128(N), K-step 64,
// double-buffered LDS filled by async copies; B-fragments software-pipelined.
// Waves 4(M) x 2(N); wave tile 32 x 64 (2x4 WMMA tiles).
// vlayout==0 : dst[b][h][s][d] (Q,K); vlayout==1 : dst[b][h][d][s] (V^T)
// ---------------------------------------------------------------------------
__global__ __launch_bounds__(256)
void qkv_proj_kernel(const unsigned short* __restrict__ xb_raw,
                     const unsigned short* __restrict__ Wb_raw,
                     const float* __restrict__ bias,
                     unsigned short* __restrict__ dst_raw,
                     int vlayout)
{
    __shared__ __bf16 lds_a[2][128 * LDT];
    __shared__ __bf16 lds_b[2][128 * LDT];

    const __bf16* xb = reinterpret_cast<const __bf16*>(xb_raw);
    const __bf16* Wb = reinterpret_cast<const __bf16*>(Wb_raw);
    __bf16* dst = reinterpret_cast<__bf16*>(dst_raw);

    const int tid  = threadIdx.x;
    const int lane = tid & 31;
    const int wave = tid >> 5;
    const int wm   = wave & 3;   // 0..3 (M)
    const int wn   = wave >> 2;  // 0..1 (N)
    const int l16  = lane & 15;
    const int half = lane >> 4;

    const int m0 = blockIdx.x * 128;
    const int n0 = blockIdx.y * 128;

    // Stage one 128x64 bf16 A tile + 128x64 B tile (16B chunks) into LDS.
    auto stage = [&](int buf, int kk) {
#pragma unroll
        for (int c = 0; c < 4; ++c) {
            const int e = tid + 256 * c;
            const int row = e >> 3, colc = e & 7;
            async_copy16(xb + (size_t)(m0 + row) * D + kk + colc * 8,
                         &lds_a[buf][row * LDT + colc * 8]);
        }
#pragma unroll
        for (int c = 0; c < 4; ++c) {
            const int e = tid + 256 * c;
            const int row = e >> 3, colc = e & 7;
            async_copy16(Wb + (size_t)(n0 + row) * D + kk + colc * 8,
                         &lds_b[buf][row * LDT + colc * 8]);
        }
    };

    f32x8 acc[2][4] = {};

    // One K-step (64) of compute from LDS buffer `buf`.
    auto compute = [&](int buf) {
        const __bf16* la = &lds_a[buf][0];
        const __bf16* lb = &lds_b[buf][0];
        Frag a[2][2];  // [kc][mi]
#pragma unroll
        for (int kc = 0; kc < 2; ++kc)
#pragma unroll
            for (int mi = 0; mi < 2; ++mi) {
                const int base = (wm * 32 + mi * 16 + l16) * LDT + kc * 32;
                a[kc][mi].h[0] = *(const bf16x8*)&la[base + half * 8];
                a[kc][mi].h[1] = *(const bf16x8*)&la[base + 16 + half * 8];
            }
        auto loadB = [&](int t, Frag& b) {
            const int kc = t >> 2, ni = t & 3;
            const int base = (wn * 64 + ni * 16 + l16) * LDT + kc * 32 + half * 8;
            b.h[0] = *(const bf16x8*)&lb[base];
            b.h[1] = *(const bf16x8*)&lb[base + 16];
        };
        Frag b[2];  // software-pipelined B fragments
        loadB(0, b[0]);
#pragma unroll
        for (int t = 0; t < 8; ++t) {
            const int kc = t >> 2, ni = t & 3;
            if (t < 7) loadB(t + 1, b[(t + 1) & 1]);
            acc[0][ni] = wmma_bf16(a[kc][0].f, b[t & 1].f, acc[0][ni]);
            acc[1][ni] = wmma_bf16(a[kc][1].f, b[t & 1].f, acc[1][ni]);
        }
    };

    stage(0, 0);
    for (int kk = 0; kk < D - 64; kk += 64) {   // steady state: always stage next
        const int cur = (kk >> 6) & 1;
        stage(cur ^ 1, kk + 64);
        WAIT_ASYNC(0x8);   // current buffer landed; next buffer's 8 still in flight
        __syncthreads();
        compute(cur);
        __syncthreads();
    }
    WAIT_ASYNC(0x0);       // last buffer
    __syncthreads();
    compute(((D - 64) >> 6) & 1);

    // bias + bf16 store (C layout: VGPR r -> row r + 8*half, N = l16)
#pragma unroll
    for (int ni = 0; ni < 4; ++ni) {
        const int n = n0 + wn * 64 + ni * 16 + l16;
        const float bv = bias[n];
        const int hh = n >> 6;
        const int dd = n & 63;
#pragma unroll
        for (int mi = 0; mi < 2; ++mi) {
            const int mbase = m0 + wm * 32 + mi * 16 + half * 8;  // 8-aligned
            if (vlayout == 0) {
#pragma unroll
                for (int r = 0; r < 8; ++r) {
                    const int m  = mbase + r;
                    const int bb = m >> 10;
                    const int ss = m & 1023;
                    dst[(((size_t)(bb * H + hh)) * S + ss) * DH + dd] =
                        (__bf16)(acc[mi][ni][r] + bv);
                }
            } else {
                // transposed layout: 8 consecutive s per lane -> one 16B store
                const int bb  = mbase >> 10;
                const int ss0 = mbase & 1023;
                bf16x8 pack;
#pragma unroll
                for (int r = 0; r < 8; ++r) pack[r] = (__bf16)(acc[mi][ni][r] + bv);
                *(bf16x8*)&dst[(((size_t)(bb * H + hh)) * DH + dd) * S + ss0] = pack;
            }
        }
    }
}

// ---------------------------------------------------------------------------
// Phase 2: flash attention. grid=(S/128, B*H), block=256 (8 waves),
// wave w owns q rows [w*16, w*16+16). K/V blocks (64 wide) async-staged into
// double-buffered LDS; online softmax; P relayout via per-wave LDS scratch.
// ---------------------------------------------------------------------------
__global__ __launch_bounds__(256)
void flash_attn_kernel(const unsigned short* __restrict__ Qh_raw,
                       const unsigned short* __restrict__ Kh_raw,
                       const unsigned short* __restrict__ Vt_raw,
                       const float* __restrict__ mask,
                       float* __restrict__ out)
{
    __shared__ __bf16 lds_k[2][64 * LDT];
    __shared__ __bf16 lds_v[2][64 * LDT];
    __shared__ __bf16 lds_p[8][16 * LDT];

    const int tid  = threadIdx.x;
    const int lane = tid & 31;
    const int wave = tid >> 5;
    const int l16  = lane & 15;
    const int half = lane >> 4;

    const int bh = blockIdx.y;
    const int b  = bh >> 4;
    const int hh = bh & 15;
    const int qbase = blockIdx.x * 128 + wave * 16;

    const __bf16* Qp = reinterpret_cast<const __bf16*>(Qh_raw) + (size_t)bh * S * DH;
    const __bf16* Kp = reinterpret_cast<const __bf16*>(Kh_raw) + (size_t)bh * S * DH;
    const __bf16* Vp = reinterpret_cast<const __bf16*>(Vt_raw) + (size_t)bh * DH * S;

    auto stage = [&](int buf, int kb) {
#pragma unroll
        for (int c = 0; c < 2; ++c) {
            const int e = tid + 256 * c;
            const int row = e >> 3, colc = e & 7;
            async_copy16(Kp + (size_t)(kb + row) * DH + colc * 8,
                         &lds_k[buf][row * LDT + colc * 8]);
        }
#pragma unroll
        for (int c = 0; c < 2; ++c) {
            const int e = tid + 256 * c;
            const int row = e >> 3, colc = e & 7;
            async_copy16(Vp + (size_t)row * S + kb + colc * 8,
                         &lds_v[buf][row * LDT + colc * 8]);
        }
    };

    // Q strip 16x64 as two A fragments, kept in registers.
    Frag aq[2];
#pragma unroll
    for (int dc = 0; dc < 2; ++dc) {
        const __bf16* p = Qp + (size_t)(qbase + l16) * DH + dc * 32 + half * 8;
        aq[dc].h[0] = *(const bf16x8*)p;
        aq[dc].h[1] = *(const bf16x8*)(p + 16);
    }

    f32x8 o[4] = {};
    float mrow[8], lrow[8];
#pragma unroll
    for (int r = 0; r < 8; ++r) { mrow[r] = -1e30f; lrow[r] = 0.0f; }

    const float scale = 0.125f;  // 1/sqrt(64)

    // One 64-wide K/V block from LDS buffer `buf`.
    auto process = [&](int buf, int kb) {
        // ---- scores: 16 x 64, K-fragments software-pipelined ----
        auto loadK = [&](int t, Frag& f) {   // t = j*2 + dc
            const int base = ((t >> 1) * 16 + l16) * LDT + (t & 1) * 32 + half * 8;
            f.h[0] = *(const bf16x8*)&lds_k[buf][base];
            f.h[1] = *(const bf16x8*)&lds_k[buf][base + 16];
        };
        Frag bk[2];
        loadK(0, bk[0]);
        f32x8 sc[4];
#pragma unroll
        for (int j = 0; j < 4; ++j) {
            f32x8 c = {};
#pragma unroll
            for (int dc = 0; dc < 2; ++dc) {
                const int t = j * 2 + dc;
                if (t < 7) loadK(t + 1, bk[(t + 1) & 1]);
                c = wmma_bf16(aq[dc].f, bk[t & 1].f, c);
            }
            const float mk   = mask[b * S + kb + j * 16 + l16];
            const float madd = (mk - 1.0f) * 10000.0f;
#pragma unroll
            for (int r = 0; r < 8; ++r) sc[j][r] = c[r] * scale + madd;
        }

        // ---- online softmax (16-lane xor reductions) ----
        float mnew[8], alpha[8];
#pragma unroll
        for (int r = 0; r < 8; ++r) {
            float v = fmaxf(fmaxf(sc[0][r], sc[1][r]), fmaxf(sc[2][r], sc[3][r]));
            v = fmaxf(v, __shfl_xor(v, 1, 32));
            v = fmaxf(v, __shfl_xor(v, 2, 32));
            v = fmaxf(v, __shfl_xor(v, 4, 32));
            v = fmaxf(v, __shfl_xor(v, 8, 32));
            mnew[r]  = fmaxf(mrow[r], v);
            alpha[r] = __expf(mrow[r] - mnew[r]);
            mrow[r]  = mnew[r];
        }
#pragma unroll
        for (int r = 0; r < 8; ++r) {
            float s = 0.0f;
#pragma unroll
            for (int j = 0; j < 4; ++j) {
                const float e = __expf(sc[j][r] - mnew[r]);
                sc[j][r] = e;
                s += e;
            }
            s += __shfl_xor(s, 1, 32);
            s += __shfl_xor(s, 2, 32);
            s += __shfl_xor(s, 4, 32);
            s += __shfl_xor(s, 8, 32);
            lrow[r] = lrow[r] * alpha[r] + s;
        }
#pragma unroll
        for (int n = 0; n < 4; ++n)
#pragma unroll
            for (int r = 0; r < 8; ++r) o[n][r] *= alpha[r];

        // ---- P: C layout -> A layout via per-wave LDS scratch ----
        __bf16* lp = &lds_p[wave][0];
#pragma unroll
        for (int j = 0; j < 4; ++j)
#pragma unroll
            for (int r = 0; r < 8; ++r)
                lp[(r + half * 8) * LDT + j * 16 + l16] = (__bf16)sc[j][r];
        __syncthreads();

        Frag pa[2];
#pragma unroll
        for (int kc = 0; kc < 2; ++kc) {
            const __bf16* p = &lds_p[wave][l16 * LDT + kc * 32 + half * 8];
            pa[kc].h[0] = *(const bf16x8*)p;
            pa[kc].h[1] = *(const bf16x8*)(p + 16);
        }

        // ---- PV from LDS-staged V^T block, V-fragments pipelined ----
        auto loadV = [&](int t, Frag& f) {   // t = n*2 + kc
            const int base = ((t >> 1) * 16 + l16) * LDT + (t & 1) * 32 + half * 8;
            f.h[0] = *(const bf16x8*)&lds_v[buf][base];
            f.h[1] = *(const bf16x8*)&lds_v[buf][base + 16];
        };
        Frag vb[2];
        loadV(0, vb[0]);
#pragma unroll
        for (int t = 0; t < 8; ++t) {
            const int n = t >> 1, kc = t & 1;
            if (t < 7) loadV(t + 1, vb[(t + 1) & 1]);
            o[n] = wmma_bf16(pa[kc].f, vb[t & 1].f, o[n]);
        }
    };

    stage(0, 0);
    for (int kb = 0; kb < S - 64; kb += 64) {
        const int cur = (kb >> 6) & 1;
        stage(cur ^ 1, kb + 64);
        WAIT_ASYNC(0x4);
        __syncthreads();
        process(cur, kb);
        __syncthreads();
    }
    WAIT_ASYNC(0x0);
    __syncthreads();
    process(((S - 64) >> 6) & 1, S - 64);

    // ---- normalize, merge heads, fp32 store ----
#pragma unroll
    for (int n = 0; n < 4; ++n) {
#pragma unroll
        for (int r = 0; r < 8; ++r) {
            const int qrow = qbase + r + half * 8;
            const float val = o[n][r] / lrow[r];
            out[((size_t)(b * S + qrow)) * D + hh * DH + n * 16 + l16] = val;
        }
    }
}

// ---------------------------------------------------------------------------
// Launcher
// ---------------------------------------------------------------------------
extern "C" void kernel_launch(void* const* d_in, const int* in_sizes, int n_in,
                              void* d_out, int out_size, void* d_ws, size_t ws_size,
                              hipStream_t stream) {
    const float* q    = (const float*)d_in[0];
    const float* k    = (const float*)d_in[1];
    const float* v    = (const float*)d_in[2];
    const float* mask = (const float*)d_in[3];
    const float* Wq   = (const float*)d_in[4];
    const float* bq   = (const float*)d_in[5];
    const float* Wk   = (const float*)d_in[6];
    const float* bk   = (const float*)d_in[7];
    const float* Wv   = (const float*)d_in[8];
    const float* bv   = (const float*)d_in[9];
    float* out = (float*)d_out;

    const size_t perX = (size_t)8 * S * D;   // 8,388,608 elems (activations / outputs)
    const size_t perW = (size_t)D * D;       // 1,048,576 elems (weights)

    unsigned short* ws = (unsigned short*)d_ws;
    unsigned short* Qh  = ws;                         // bf16 [B,H,S,DH]
    unsigned short* Kh  = Qh  + perX;
    unsigned short* Vt  = Kh  + perX;                 // bf16 [B,H,DH,S]
    unsigned short* qb  = Vt  + perX;                 // bf16 activations
    unsigned short* kb2 = qb  + perX;
    unsigned short* vb2 = kb2 + perX;
    unsigned short* Wqb = vb2 + perX;                 // bf16 weights
    unsigned short* Wkb = Wqb + perW;
    unsigned short* Wvb = Wkb + perW;

    dim3 blk(256);
    {   // fp32 -> bf16 converts
        const int nx4 = (int)(perX / 4), nw4 = (int)(perW / 4);
        dim3 gx((nx4 + 255) / 256), gw((nw4 + 255) / 256);
        cvt_f32_to_bf16_kernel<<<gx, blk, 0, stream>>>(q, qb, nx4);
        cvt_f32_to_bf16_kernel<<<gx, blk, 0, stream>>>(k, kb2, nx4);
        cvt_f32_to_bf16_kernel<<<gx, blk, 0, stream>>>(v, vb2, nx4);
        cvt_f32_to_bf16_kernel<<<gw, blk, 0, stream>>>(Wq, Wqb, nw4);
        cvt_f32_to_bf16_kernel<<<gw, blk, 0, stream>>>(Wk, Wkb, nw4);
        cvt_f32_to_bf16_kernel<<<gw, blk, 0, stream>>>(Wv, Wvb, nw4);
    }

    dim3 g1(64, 8);  // M/128 x N/128
    qkv_proj_kernel<<<g1, blk, 0, stream>>>(qb,  Wqb, bq, Qh, 0);
    qkv_proj_kernel<<<g1, blk, 0, stream>>>(kb2, Wkb, bk, Kh, 0);
    qkv_proj_kernel<<<g1, blk, 0, stream>>>(vb2, Wvb, bv, Vt, 1);

    dim3 g2(8, 8 * H);  // S/128 x B*H
    flash_attn_kernel<<<g2, blk, 0, stream>>>(Qh, Kh, Vt, mask, out);
}
